// QuantumHybridResNet_87797721465144
// MI455X (gfx1250) — matrix operand order repaired and checked by reading, not compiled, not run on hardware
//
#include <hip/hip_runtime.h>
#include <hip/hip_bf16.h>

// ---------------------------------------------------------------------------
// MI455X (gfx1250) hybrid ResNet18 + 4-qubit circuit forward pass.
// Convs: implicit GEMM on v_wmma_f32_16x16x32_f16 (wave32), NHWC f16.
//  - NHWC activations: a 64-wide K-tile = ONE filter tap, c contiguous
//    -> A staging is uint4 vector copies, bounds check per row not per element
//  - f16 end-to-end (x & weights repacked once); epilogue index = p*Cout+oc
//  - double-buffered LDS, ONE barrier per 64-wide K-step, 4 WMMA / step
//  - BN(running stats) + residual + ReLU fused into the epilogue
// ---------------------------------------------------------------------------

typedef _Float16 f16;
typedef f16  v16h __attribute__((ext_vector_type(16)));
typedef float v8f  __attribute__((ext_vector_type(8)));

#define PI_F 3.14159265358979f
#define BATCH 64

// ---------------------------------------------------------------------------
// one-time repacks
// ---------------------------------------------------------------------------
__global__ void x_nchw_to_nhwc(const float* __restrict__ in, f16* __restrict__ out)
{
  constexpr int N = BATCH, C = 3, H = 224, W = 224;
  int i = blockIdx.x * blockDim.x + threadIdx.x;
  if (i >= N * H * W * C) return;
  int c = i % C; int t = i / C;
  int w = t % W; t /= W;
  int h = t % H; int n = t / H;
  out[i] = (f16)in[((n * C + c) * H + h) * W + w];
}

// OIHW f32 -> [oc][(r,s,c)] f16
__global__ void wrepack_kernel(const float* __restrict__ in, f16* __restrict__ out,
                               int Cout, int C, int R, int S)
{
  int t = blockIdx.x * blockDim.x + threadIdx.x;
  if (t >= Cout * R * S * C) return;
  int c = t % C; int u = t / C;
  int s = u % S; u /= S;
  int r = u % R; int oc = u / R;
  out[t] = (f16)in[((oc * C + c) * R + r) * S + s];
}

// ---------------------------------------------------------------------------
// conv + BN (+residual)(+relu), implicit GEMM, NHWC f16.
// Block tile 64 pixels x 64 oc, 8 waves (4Mx2N), K stepped by 64.
// ---------------------------------------------------------------------------
template <int C, int H, int W, int Cout, int R, int S, int STRIDE, int PAD>
__launch_bounds__(256)
__global__ void conv_bn_wmma(
    const f16* __restrict__ x, const f16* __restrict__ wgt,
    const float* __restrict__ bnb, const float* __restrict__ bng,
    const float* __restrict__ bnm, const float* __restrict__ bnv,
    const f16* __restrict__ resid, f16* __restrict__ out, int relu)
{
  constexpr int HO   = (H + 2 * PAD - R) / STRIDE + 1;
  constexpr int WO   = (W + 2 * PAD - S) / STRIDE + 1;
  constexpr int HOWO = HO * WO;
  constexpr int Ktot = C * R * S;                 // k ordered (r,s,c)
  constexpr int KT   = (Ktot + 63) / 64;
  constexpr bool FAST = (C % 64 == 0);            // every layer but the stem

  __shared__ f16 lA[2][64][72];   // 64 pixels x 64 K, +8 pad (144B rows, 16B-aligned)
  __shared__ f16 lB[2][64][72];   // 64 oc x 64 K

  const int tid  = threadIdx.x;
  const int lane = tid & 31;
  const int wave = tid >> 5;
  const int mw   = wave >> 1;
  const int nw   = wave & 1;

  const int blockM = blockIdx.x * 64;
  const int blockN = blockIdx.y * 64;

  // FAST staging geometry: 8 threads x 16B per row; rows rA, rA+32
  const int tc = tid & 7;
  const int rA = tid >> 3;
  // stem staging geometry: column kk, rows row0+4i
  const int kk   = tid & 63;
  const int row0 = tid >> 6;

  int g2N[2], g2H[2], g2W[2];          // FAST: 2 hoisted pixel decodes
  int g16N[16], g16H[16], g16W[16];    // stem: 16 hoisted pixel decodes
  if constexpr (FAST) {
    #pragma unroll
    for (int j = 0; j < 2; ++j) {
      int p  = blockM + rA + 32 * j;
      int n  = p / HOWO;
      int rp = p - n * HOWO;
      int ho = rp / WO;
      int wo = rp - ho * WO;
      g2N[j] = n * (H * W * C);
      g2H[j] = ho * STRIDE - PAD;
      g2W[j] = wo * STRIDE - PAD;
    }
  } else {
    #pragma unroll
    for (int i = 0; i < 16; ++i) {
      int p  = blockM + row0 + 4 * i;
      int n  = p / HOWO;
      int rp = p - n * HOWO;
      int ho = rp / WO;
      int wo = rp - ho * WO;
      g16N[i] = n * (H * W * C);
      g16H[i] = ho * STRIDE - PAD;
      g16W[i] = wo * STRIDE - PAD;
    }
  }
  const int wBase = (blockN + row0) * Ktot + kk;   // stem B base

  auto stage = [&](int buf, int kbase) {
    if constexpr (FAST) {
      // one filter tap per 64-wide K-tile: decode once
      int rs = kbase / C;
      int cb = kbase - rs * C;
      int r  = rs / S;
      int s  = rs - r * S;
      #pragma unroll
      for (int j = 0; j < 2; ++j) {
        int hi = g2H[j] + r, wi = g2W[j] + s;
        bool ok = true;
        if constexpr (PAD > 0)
          ok = ((unsigned)hi < (unsigned)H) && ((unsigned)wi < (unsigned)W);
        int idx = g2N[j] + (hi * W + wi) * C + cb + tc * 8;
        uint4 d = *(const uint4*)&x[ok ? idx : 0];
        if constexpr (PAD > 0) {
          d.x = ok ? d.x : 0u; d.y = ok ? d.y : 0u;
          d.z = ok ? d.z : 0u; d.w = ok ? d.w : 0u;
        }
        *(uint4*)&lA[buf][rA + 32 * j][tc * 8] = d;
      }
      #pragma unroll
      for (int j = 0; j < 2; ++j) {
        int oc = rA + 32 * j;
        uint4 d = *(const uint4*)&wgt[(blockN + oc) * Ktot + kbase + tc * 8];
        *(uint4*)&lB[buf][oc][tc * 8] = d;
      }
    } else {
      // stem: C=3, K=147, scalar guarded path
      int kg  = kbase + kk;
      bool kin = (kg < Ktot);
      int rs = kg / 3;
      int c  = kg - 3 * rs;
      int r  = rs / S;
      int s  = rs - r * S;
      #pragma unroll
      for (int i = 0; i < 16; ++i) {
        int hi = g16H[i] + r, wi = g16W[i] + s;
        bool ok = kin && ((unsigned)hi < (unsigned)H) && ((unsigned)wi < (unsigned)W);
        int idx = g16N[i] + (hi * W + wi) * C + c;
        f16 v = x[ok ? idx : 0];
        lA[buf][row0 + 4 * i][kk] = ok ? v : (f16)0.f;
      }
      #pragma unroll
      for (int i = 0; i < 16; ++i) {
        int idx = wBase + i * (4 * Ktot) + kbase;
        f16 v = wgt[kin ? idx : 0];
        lB[buf][row0 + 4 * i][kk] = kin ? v : (f16)0.f;
      }
    }
  };

  // fragment geometry (m/n = lane%16; contiguous k = 16*(lane/16) + s for A and B)
  const int mrow = (mw << 4) + (lane & 15);
  const int kh   = (lane >> 4) << 3;
  const int kb   = (lane >> 4) << 4;
  const int nn0  = (nw << 5) + (lane & 15);
  const int nn1  = nn0 + 16;

  v8f acc0 = {0.f,0.f,0.f,0.f,0.f,0.f,0.f,0.f};
  v8f acc1 = {0.f,0.f,0.f,0.f,0.f,0.f,0.f,0.f};

  stage(0, 0);

  for (int kt = 0; kt < KT; ++kt) {
    __syncthreads();
    const int buf = kt & 1;
    if (kt + 1 < KT) {
      __builtin_prefetch(&wgt[(blockN + rA) * Ktot + (kt + 1) * 64], 0, 0);
      stage(1 - buf, (kt + 1) * 64);
    }
    #pragma unroll
    for (int ks = 0; ks < 64; ks += 32) {
      union { v16h v; unsigned u[8]; } fa, fb0, fb1;
      #pragma unroll
      for (int j = 0; j < 8; ++j) {
        fa.u[j]     = *(const unsigned*)&lA[buf][mrow][ks + kb + 2 * j];
      }
      #pragma unroll
      for (int j = 0; j < 8; ++j) {
        fb0.u[j] = *(const unsigned*)&lB[buf][nn0][ks + kb + 2 * j];
        fb1.u[j] = *(const unsigned*)&lB[buf][nn1][ks + kb + 2 * j];
      }
      acc0 = __builtin_amdgcn_wmma_f32_16x16x32_f16(false, fa.v, false, fb0.v,
                                                    (short)0, acc0, false, false);
      acc1 = __builtin_amdgcn_wmma_f32_16x16x32_f16(false, fa.v, false, fb1.v,
                                                    (short)0, acc1, false, false);
    }
  }

  // ---- fused epilogue (NHWC: index = pixel*Cout + oc, no division) ----
  const int nl  = lane & 15;
  const int oc0 = blockN + (nw << 5) + nl;
  const int oc1 = oc0 + 16;
  const float sc0 = bng[oc0] * rsqrtf(bnv[oc0] + 1e-5f);
  const float bs0 = bnb[oc0] - bnm[oc0] * sc0;
  const float sc1 = bng[oc1] * rsqrtf(bnv[oc1] + 1e-5f);
  const float bs1 = bnb[oc1] - bnm[oc1] * sc1;
  const int rowoff = (lane >> 4) << 3;

  #pragma unroll
  for (int r8 = 0; r8 < 8; ++r8) {
    int p  = blockM + (mw << 4) + rowoff + r8;
    int o0 = p * Cout + oc0;
    int o1 = p * Cout + oc1;
    float v0 = acc0[r8] * sc0 + bs0;
    float v1 = acc1[r8] * sc1 + bs1;
    if (resid) { v0 += (float)resid[o0]; v1 += (float)resid[o1]; }
    if (relu)  { v0 = v0 > 0.f ? v0 : 0.f; v1 = v1 > 0.f ? v1 : 0.f; }
    out[o0] = (f16)v0;
    out[o1] = (f16)v1;
  }
}

// ---------------------------------------------------------------------------
// 3x3/2 pad1 max pool, NHWC f16: (64,112,112,64) -> (64,56,56,64)
// ---------------------------------------------------------------------------
__global__ void maxpool_kernel(const f16* __restrict__ in, f16* __restrict__ out)
{
  constexpr int C = 64, H = 112, W = 112, HO = 56, WO = 56;
  int idx = blockIdx.x * blockDim.x + threadIdx.x;
  if (idx >= BATCH * HO * WO * C) return;
  int c  = idx & 63; int t = idx >> 6;
  int wo = t % WO;   t /= WO;
  int ho = t % HO;   int n = t / HO;
  float m = -3.402823466e38f;
  #pragma unroll
  for (int r = 0; r < 3; ++r)
    #pragma unroll
    for (int s = 0; s < 3; ++s) {
      int hi = ho * 2 - 1 + r, wi = wo * 2 - 1 + s;
      if (hi >= 0 && hi < H && wi >= 0 && wi < W) {
        float v = (float)in[((n * H + hi) * W + wi) * C + c];
        m = v > m ? v : m;
      }
    }
  out[idx] = (f16)m;
}

// ---------------------------------------------------------------------------
// global average pool, NHWC f16 (64,7,7,512) -> f32 (64,512)
// ---------------------------------------------------------------------------
__global__ void avgpool_kernel(const f16* __restrict__ in, float* __restrict__ out)
{
  int idx = blockIdx.x * blockDim.x + threadIdx.x;
  if (idx >= BATCH * 512) return;
  int c = idx & 511; int n = idx >> 9;
  float s = 0.f;
  #pragma unroll
  for (int p = 0; p < 49; ++p) s += (float)in[(n * 49 + p) * 512 + c];
  out[idx] = s * (1.f / 49.f);
}

// ---------------------------------------------------------------------------
// bottleneck: z = tanh(feat @ W^T + b) * pi   (64 x 512 -> 64 x 4)
// ---------------------------------------------------------------------------
__global__ void bottleneck_kernel(const float* __restrict__ feat, const float* __restrict__ Wm,
                                  const float* __restrict__ b, float* __restrict__ z)
{
  int t = blockIdx.x * blockDim.x + threadIdx.x;
  if (t >= BATCH * 4) return;
  int bi = t >> 2, q = t & 3;
  float s = b[q];
  #pragma unroll 1
  for (int i = 0; i < 512; ++i) s += feat[bi * 512 + i] * Wm[q * 512 + i];
  z[t] = tanhf(s) * PI_F;
}

// ---------------------------------------------------------------------------
// 4-qubit circuit + post linear, one thread per sample.
// ---------------------------------------------------------------------------
__global__ void quantum_kernel(const float* __restrict__ z, const float* __restrict__ qw,
                               const float* __restrict__ pW, const float* __restrict__ pb,
                               float* __restrict__ out)
{
  int b = blockIdx.x * blockDim.x + threadIdx.x;
  if (b >= BATCH) return;

  float sr[16], si[16];
  #pragma unroll
  for (int i = 0; i < 16; ++i) { sr[i] = 0.f; si[i] = 0.f; }
  sr[0] = 1.f;

  auto gate1 = [&](int w, float u00r, float u00i, float u01r, float u01i,
                          float u10r, float u10i, float u11r, float u11i) {
    int bit = 8 >> w;
    #pragma unroll
    for (int i = 0; i < 16; ++i) {
      if (!(i & bit)) {
        int j = i | bit;
        float ar = sr[i], ai = si[i], br = sr[j], bi_ = si[j];
        sr[i] = u00r * ar - u00i * ai + u01r * br - u01i * bi_;
        si[i] = u00r * ai + u00i * ar + u01r * bi_ + u01i * br;
        sr[j] = u10r * ar - u10i * ai + u11r * br - u11i * bi_;
        si[j] = u10r * ai + u10i * ar + u11r * bi_ + u11i * br;
      }
    }
  };

  #pragma unroll 1
  for (int w = 0; w < 4; ++w) {                 // AngleEmbedding RX
    float th = z[b * 4 + w];
    float c = cosf(th * 0.5f), s = sinf(th * 0.5f);
    gate1(w, c, 0.f, 0.f, -s, 0.f, -s, c, 0.f);
  }
  #pragma unroll 1
  for (int l = 0; l < 2; ++l) {                 // StronglyEntanglingLayers
    #pragma unroll 1
    for (int w = 0; w < 4; ++w) {
      float phi = qw[(l * 4 + w) * 3 + 0];
      float th  = qw[(l * 4 + w) * 3 + 1];
      float om  = qw[(l * 4 + w) * 3 + 2];
      float c = cosf(th * 0.5f), s = sinf(th * 0.5f);
      float a = 0.5f * (phi + om), d = 0.5f * (phi - om);
      gate1(w,  cosf(a) * c, -sinf(a) * c,
               -cosf(d) * s, -sinf(d) * s,
                cosf(d) * s, -sinf(d) * s,
                cosf(a) * c,  sinf(a) * c);
    }
    int r = (l % 3) + 1;
    #pragma unroll 1
    for (int w = 0; w < 4; ++w) {
      int cb = 8 >> w, tb = 8 >> ((w + r) & 3);
      #pragma unroll
      for (int i = 0; i < 16; ++i) {
        if ((i & cb) && !(i & tb)) {
          int j = i | tb;
          float tr = sr[i], ti = si[i];
          sr[i] = sr[j]; si[i] = si[j];
          sr[j] = tr;    si[j] = ti;
        }
      }
    }
  }
  float res = pb[0];
  #pragma unroll 1
  for (int q = 0; q < 4; ++q) {
    int bit = 8 >> q;
    float e = 0.f;
    #pragma unroll
    for (int i = 0; i < 16; ++i) {
      float p = sr[i] * sr[i] + si[i] * si[i];
      e += (i & bit) ? -p : p;
    }
    res += e * pW[q];
  }
  out[b] = res;
}

// ---------------------------------------------------------------------------
// Host-side templated launcher
// ---------------------------------------------------------------------------
template <int C, int H, int W, int Cout, int R, int S, int STRIDE, int PAD>
static void launch_conv(const f16* x, const f16* w,
                        const float* bnb, const float* bng,
                        const float* bnm, const float* bnv,
                        const f16* resid, f16* out, int relu, hipStream_t stream)
{
  constexpr int HO = (H + 2 * PAD - R) / STRIDE + 1;
  constexpr int WO = (W + 2 * PAD - S) / STRIDE + 1;
  constexpr int M  = BATCH * HO * WO;
  dim3 grid(M / 64, Cout / 64);
  conv_bn_wmma<C, H, W, Cout, R, S, STRIDE, PAD><<<grid, 256, 0, stream>>>(
      x, w, bnb, bng, bnm, bnv, resid, out, relu);
}

// ---------------------------------------------------------------------------
// Host orchestration. Inputs flattened in jax tree order (dict keys sorted):
// params{backbone{bn1(beta,gamma,mean,var), conv1, layer1..4[blocks:
// bn1,bn2,conv1,conv2,(down_bn,down_conv)]}, bottleneck{W,b}, post{W,b},
// q_weights}, then x.
// ---------------------------------------------------------------------------
extern "C" void kernel_launch(void* const* d_in, const int* in_sizes, int n_in,
                              void* d_out, int out_size, void* d_ws, size_t ws_size,
                              hipStream_t stream)
{
  (void)in_sizes; (void)n_in; (void)out_size; (void)ws_size;

  auto IN = [&](int i) { return (const float*)d_in[i]; };

  size_t off = 0;
  auto alloc = [&](size_t bytes) -> void* {
    void* p = (char*)d_ws + off;
    off += (bytes + 255) & ~(size_t)255;
    return p;
  };

  // f16 NHWC activation buffers
  f16* conv1_out = (f16*)alloc((size_t)BATCH * 112 * 112 * 64 * 2);
  f16* P0   = (f16*)alloc((size_t)BATCH * 56 * 56 * 64 * 2);
  f16* P1   = (f16*)alloc((size_t)BATCH * 56 * 56 * 64 * 2);
  f16* P2   = (f16*)alloc((size_t)BATCH * 56 * 56 * 64 * 2);
  f16* Rbuf = (f16*)alloc((size_t)BATCH * 28 * 28 * 128 * 2);
  float* feat = (float*)alloc((size_t)BATCH * 512 * 4);
  float* zbuf = (float*)alloc((size_t)BATCH * 4 * 4);

  // one-time repacks: x NCHW f32 -> NHWC f16; weights OIHW f32 -> [oc][(r,s,c)] f16
  f16* xh = (f16*)alloc((size_t)BATCH * 224 * 224 * 3 * 2);
  {
    int n = BATCH * 224 * 224 * 3;
    x_nchw_to_nhwc<<<(n + 255) / 256, 256, 0, stream>>>(IN(105), xh);
  }
  auto wrep = [&](int idx, int Cout, int C, int R, int S) -> f16* {
    int n = Cout * R * S * C;
    f16* p = (f16*)alloc((size_t)n * 2);
    wrepack_kernel<<<(n + 255) / 256, 256, 0, stream>>>(IN(idx), p, Cout, C, R, S);
    return p;
  };
  f16* w4  = wrep(4,   64,   3, 7, 7);
  f16* w13 = wrep(13,  64,  64, 3, 3);
  f16* w14 = wrep(14,  64,  64, 3, 3);
  f16* w23 = wrep(23,  64,  64, 3, 3);
  f16* w24 = wrep(24,  64,  64, 3, 3);
  f16* w33 = wrep(33, 128,  64, 3, 3);
  f16* w39 = wrep(39, 128,  64, 1, 1);
  f16* w34 = wrep(34, 128, 128, 3, 3);
  f16* w48 = wrep(48, 128, 128, 3, 3);
  f16* w49 = wrep(49, 128, 128, 3, 3);
  f16* w58 = wrep(58, 256, 128, 3, 3);
  f16* w64 = wrep(64, 256, 128, 1, 1);
  f16* w59 = wrep(59, 256, 256, 3, 3);
  f16* w73 = wrep(73, 256, 256, 3, 3);
  f16* w74 = wrep(74, 256, 256, 3, 3);
  f16* w83 = wrep(83, 512, 256, 3, 3);
  f16* w89 = wrep(89, 512, 256, 1, 1);
  f16* w84 = wrep(84, 512, 512, 3, 3);
  f16* w98 = wrep(98, 512, 512, 3, 3);
  f16* w99 = wrep(99, 512, 512, 3, 3);

  #define BN(base) IN(base + 0), IN(base + 1), IN(base + 2), IN(base + 3)

  // stem: conv1 7x7/2 pad3 + bn + relu, then 3x3/2 maxpool
  launch_conv<3, 224, 224, 64, 7, 7, 2, 3>(xh, w4, BN(0), nullptr, conv1_out, 1, stream);
  {
    int total = BATCH * 56 * 56 * 64;
    maxpool_kernel<<<(total + 255) / 256, 256, 0, stream>>>(conv1_out, P0);
  }

  f16* bufs[3] = {P0, P1, P2};
  int cur = 0;
  auto rot = [&](int k) { return bufs[(cur + k) % 3]; };

  // ---- layer1 (64ch, 56x56) ----
  launch_conv<64, 56, 56, 64, 3, 3, 1, 1>(rot(0), w13, BN(5), nullptr, rot(1), 1, stream);
  launch_conv<64, 56, 56, 64, 3, 3, 1, 1>(rot(1), w14, BN(9), rot(0), rot(2), 1, stream);
  cur = (cur + 2) % 3;
  launch_conv<64, 56, 56, 64, 3, 3, 1, 1>(rot(0), w23, BN(15), nullptr, rot(1), 1, stream);
  launch_conv<64, 56, 56, 64, 3, 3, 1, 1>(rot(1), w24, BN(19), rot(0), rot(2), 1, stream);
  cur = (cur + 2) % 3;

  // ---- layer2 (128ch, 28x28) ----
  launch_conv<64, 56, 56, 128, 3, 3, 2, 1>(rot(0), w33, BN(25), nullptr, rot(1), 1, stream);
  launch_conv<64, 56, 56, 128, 1, 1, 2, 0>(rot(0), w39, BN(35), nullptr, Rbuf, 0, stream);
  launch_conv<128, 28, 28, 128, 3, 3, 1, 1>(rot(1), w34, BN(29), Rbuf, rot(2), 1, stream);
  cur = (cur + 2) % 3;
  launch_conv<128, 28, 28, 128, 3, 3, 1, 1>(rot(0), w48, BN(40), nullptr, rot(1), 1, stream);
  launch_conv<128, 28, 28, 128, 3, 3, 1, 1>(rot(1), w49, BN(44), rot(0), rot(2), 1, stream);
  cur = (cur + 2) % 3;

  // ---- layer3 (256ch, 14x14) ----
  launch_conv<128, 28, 28, 256, 3, 3, 2, 1>(rot(0), w58, BN(50), nullptr, rot(1), 1, stream);
  launch_conv<128, 28, 28, 256, 1, 1, 2, 0>(rot(0), w64, BN(60), nullptr, Rbuf, 0, stream);
  launch_conv<256, 14, 14, 256, 3, 3, 1, 1>(rot(1), w59, BN(54), Rbuf, rot(2), 1, stream);
  cur = (cur + 2) % 3;
  launch_conv<256, 14, 14, 256, 3, 3, 1, 1>(rot(0), w73, BN(65), nullptr, rot(1), 1, stream);
  launch_conv<256, 14, 14, 256, 3, 3, 1, 1>(rot(1), w74, BN(69), rot(0), rot(2), 1, stream);
  cur = (cur + 2) % 3;

  // ---- layer4 (512ch, 7x7) ----
  launch_conv<256, 14, 14, 512, 3, 3, 2, 1>(rot(0), w83, BN(75), nullptr, rot(1), 1, stream);
  launch_conv<256, 14, 14, 512, 1, 1, 2, 0>(rot(0), w89, BN(85), nullptr, Rbuf, 0, stream);
  launch_conv<512, 7, 7, 512, 3, 3, 1, 1>(rot(1), w84, BN(79), Rbuf, rot(2), 1, stream);
  cur = (cur + 2) % 3;
  launch_conv<512, 7, 7, 512, 3, 3, 1, 1>(rot(0), w98, BN(90), nullptr, rot(1), 1, stream);
  launch_conv<512, 7, 7, 512, 3, 3, 1, 1>(rot(1), w99, BN(94), rot(0), rot(2), 1, stream);
  cur = (cur + 2) % 3;

  // head: avgpool -> bottleneck(tanh*pi) -> quantum circuit -> post linear
  avgpool_kernel<<<(BATCH * 512 + 255) / 256, 256, 0, stream>>>(bufs[cur], feat);
  bottleneck_kernel<<<1, 256, 0, stream>>>(feat, IN(100), IN(101), zbuf);
  quantum_kernel<<<2, 32, 0, stream>>>(zbuf, IN(104), IN(102), IN(103), (float*)d_out);

  #undef BN
}
